// TransformerBlockHybrid_65481071401488
// MI455X (gfx1250) — matrix operand order repaired and checked by reading, not compiled, not run on hardware
//
#include <hip/hip_runtime.h>

#define DEV __device__ __forceinline__

typedef unsigned int   u32;
typedef unsigned short u16;
typedef __attribute__((ext_vector_type(16))) __bf16 v16bf;
typedef __attribute__((ext_vector_type(8)))  float  v8f;
typedef __attribute__((ext_vector_type(4)))  u32    u32x4;
typedef __attribute__((ext_vector_type(2)))  u32    u32x2;
typedef __attribute__((ext_vector_type(4)))  float  f32x4;

union Frag { u32x4 q[2]; v16bf v; };

constexpr int Bb = 4, Ss = 2048, Ee = 1024, Hh = 16, Ff = 4096;

// f32 -> bf16 round-to-nearest-even
DEV u16 bfb(float f) {
  u32 a = __float_as_uint(f);
  a += 0x7FFFu + ((a >> 16) & 1u);
  return (u16)(a >> 16);
}
DEV u32 pk2(float lo, float hi) {
  u32 a = __float_as_uint(lo), b = __float_as_uint(hi);
  a += 0x7FFFu + ((a >> 16) & 1u);
  b += 0x7FFFu + ((b >> 16) & 1u);
  return (a >> 16) | (b & 0xFFFF0000u);
}

DEV v8f wmma_bf16(const Frag& a, const Frag& b, v8f c) {
  return __builtin_amdgcn_wmma_f32_16x16x32_bf16(false, a.v, false, b.v,
                                                 (short)0, c, false, false);
}

DEV void wave_lds_sync() {
  __builtin_amdgcn_wave_barrier();
  asm volatile("s_wait_dscnt 0x0" ::: "memory");
  __builtin_amdgcn_wave_barrier();
}

// Async global -> LDS copy, 16B per lane. Tracked by ASYNCcnt.
DEV void async_ld_b128(u32 lds_byte_off, const void* gaddr) {
  asm volatile("global_load_async_to_lds_b128 %0, %1, off"
               :: "v"(lds_byte_off), "v"(gaddr) : "memory");
}
DEV void wait_async0() {
  asm volatile("s_wait_asynccnt 0x0" ::: "memory");
}

// ---------------------------------------------------------------------------
// One-pass f32 -> packed bf16 conversion. n must be a multiple of 2048.
// ---------------------------------------------------------------------------
__launch_bounds__(256)
__global__ void f32_to_bf16(const float* __restrict__ in, u32* __restrict__ out)
{
  const size_t i = (size_t)blockIdx.x * 256 + threadIdx.x;
  f32x4 a = *(const f32x4*)(in + i * 8);
  f32x4 b = *(const f32x4*)(in + i * 8 + 4);
  u32x4 o = {pk2(a.x, a.y), pk2(a.z, a.w), pk2(b.x, b.y), pk2(b.z, b.w)};
  *(u32x4*)(out + i * 4) = o;
}

// ---------------------------------------------------------------------------
// GEMM: C[M,N] = act(A[M,K] @ W[N,K]^T + bias[N]), A/W packed bf16.
// Output: f32 (Cf) or packed bf16 (Cb). 256 thr = 8 waves (2x4),
// block tile 128x128, wave tile 64x32, k-step 64 -> 16 WMMA/wave/k-step.
// Tiles staged with GLOBAL_LOAD_ASYNC_TO_LDS_B128 (ASYNCcnt).
// ---------------------------------------------------------------------------
__launch_bounds__(256)
__global__ void gemm_bf16(const u32* __restrict__ Abf,   // [M, K/2]
                          const u32* __restrict__ Wbf,   // [N, K/2]
                          const float* __restrict__ bias,
                          float* __restrict__ Cf,
                          u16* __restrict__ Cb,
                          int M, int N, int K, int relu)
{
  __shared__ u32 lA[128 * 32];   // 128 rows x 64 bf16 = 16 KB
  __shared__ u32 lW[128 * 32];

  const int tid   = threadIdx.x;
  const int lane  = tid & 31;
  const int wave  = tid >> 5;
  const int waveM = wave >> 2;       // 0..1  (64 rows each)
  const int waveN = wave & 3;        // 0..3  (32 cols each)
  const int blockM = blockIdx.y * 128;
  const int blockN = blockIdx.x * 128;
  const int row16 = lane & 15;
  const int khalf = lane >> 4;
  const int K2 = K >> 1;             // row stride in u32

  const u32 ldsA = (u32)(size_t)(&lA[0]);
  const u32 ldsW = (u32)(size_t)(&lW[0]);

  v8f acc[4][2];
  #pragma unroll
  for (int i = 0; i < 4; ++i)
    #pragma unroll
    for (int j = 0; j < 2; ++j)
      acc[i][j] = (v8f){0,0,0,0,0,0,0,0};

  for (int k0 = 0; k0 < K; k0 += 64) {
    const int k2 = k0 >> 1;
    #pragma unroll
    for (int i = 0; i < 2; ++i) {            // A: 512 x 16B
      int e = tid + 256 * i;
      int r = e >> 3, cj = e & 7;
      async_ld_b128(ldsA + (u32)e * 16,
                    Abf + (size_t)(blockM + r) * K2 + k2 + cj * 4);
    }
    #pragma unroll
    for (int i = 0; i < 2; ++i) {            // W: 512 x 16B
      int e = tid + 256 * i;
      int r = e >> 3, cj = e & 7;
      async_ld_b128(ldsW + (u32)e * 16,
                    Wbf + (size_t)(blockN + r) * K2 + k2 + cj * 4);
    }
    if (k0 + 64 < K) {                       // hint next K tile toward caches
      __builtin_prefetch(Abf + (size_t)(blockM + (tid >> 1)) * K2 + k2 + 32 + (tid & 1) * 16, 0, 1);
      __builtin_prefetch(Wbf + (size_t)(blockN + (tid >> 1)) * K2 + k2 + 32 + (tid & 1) * 16, 0, 1);
    }
    wait_async0();
    __syncthreads();

    #pragma unroll
    for (int kc = 0; kc < 2; ++kc) {         // two 32-K chunks
      Frag a[4], b[2];
      #pragma unroll
      for (int mi = 0; mi < 4; ++mi) {
        const u32x4* p = (const u32x4*)&lA[(waveM * 64 + mi * 16 + row16) * 32];
        a[mi].q[0] = p[kc * 4 + khalf];      // K 0-7 / 8-15 of chunk
        a[mi].q[1] = p[kc * 4 + 2 + khalf];  // K 16-23 / 24-31
      }
      #pragma unroll
      for (int ni = 0; ni < 2; ++ni) {
        const u32x4* p = (const u32x4*)&lW[(waveN * 32 + ni * 16 + row16) * 32];
        b[ni].q[0] = p[kc * 4 + khalf];      // K 0-7 / 8-15 of chunk
        b[ni].q[1] = p[kc * 4 + 2 + khalf];  // K 16-23 / 24-31
      }
      #pragma unroll
      for (int mi = 0; mi < 4; ++mi)
        #pragma unroll
        for (int ni = 0; ni < 2; ++ni)
          acc[mi][ni] = wmma_bf16(a[mi], b[ni], acc[mi][ni]);
    }
    __syncthreads();
  }

  #pragma unroll
  for (int mi = 0; mi < 4; ++mi) {
    int row0 = blockM + waveM * 64 + mi * 16 + 8 * khalf;
    #pragma unroll
    for (int ni = 0; ni < 2; ++ni) {
      int col = blockN + waveN * 32 + ni * 16 + row16;
      float bi = bias[col];
      #pragma unroll
      for (int r = 0; r < 8; ++r) {
        float v = acc[mi][ni][r] + bi;
        if (relu) v = fmaxf(v, 0.0f);
        if (Cf) Cf[(size_t)(row0 + r) * N + col] = v;
        else    Cb[(size_t)(row0 + r) * N + col] = bfb(v);
      }
    }
  }
}

// ---------------------------------------------------------------------------
// Flash attention over packed-bf16 qkv [B,S,3E]; bf16 output o [B,S,E].
// Block = (b,h) x 128 queries, wave = 16 queries.
// S^T = K @ Q^T (softmax stats per-lane), O^T = V^T @ P^T (rescale per-lane).
// ---------------------------------------------------------------------------
__launch_bounds__(256)
__global__ void attn_fwd(const u16* __restrict__ qkv, u16* __restrict__ o)
{
  __shared__ u16 vT[64 * 32];        // V chunk transposed: [dk][key]
  __shared__ u16 pT[8 * 16 * 32];    // per-wave P: [q][key]

  const int tid  = threadIdx.x;
  const int lane = tid & 31;
  const int wave = tid >> 5;
  const int bh = blockIdx.x;
  const int b = bh >> 4, h = bh & 15;
  const int qblk = blockIdx.y;
  const int ql = lane & 15;
  const int khalf = lane >> 4;
  const float scale = 0.125f;        // 1/sqrt(64)

  const size_t rowQ = (size_t)(b * Ss + qblk * 128 + wave * 16 + ql) * (3 * Ee);

  Frag qf[2];
  #pragma unroll
  for (int c = 0; c < 2; ++c) {
    const u32x4* p = (const u32x4*)(qkv + rowQ + h * 64 + c * 32 + khalf * 8);
    qf[c].q[0] = p[0];
    qf[c].q[1] = p[2];
  }

  v8f oacc[4];
  #pragma unroll
  for (int d = 0; d < 4; ++d) oacc[d] = (v8f){0,0,0,0,0,0,0,0};
  float m_run = -1e30f, l_run = 0.0f;

  const int vkey = tid >> 3;          // 0..31
  const int vdk  = (tid & 7) * 8;     // 0..56
  u16* pMy = &pT[wave * 512];

  for (int kc = 0; kc < Ss / 32; ++kc) {
    const int kbase = kc * 32;
    __syncthreads();
    { // stage V chunk (32 keys x 64 dk) transposed into LDS
      u32x4 vv = *(const u32x4*)(qkv + (size_t)(b * Ss + kbase + vkey) * (3 * Ee)
                                 + 2 * Ee + h * 64 + vdk);
      #pragma unroll
      for (int j = 0; j < 4; ++j) {
        u32 w = vv[j];
        vT[(vdk + 2 * j + 0) * 32 + vkey] = (u16)(w & 0xFFFFu);
        vT[(vdk + 2 * j + 1) * 32 + vkey] = (u16)(w >> 16);
      }
    }
    __syncthreads();

    v8f st[2] = {(v8f){0,0,0,0,0,0,0,0}, (v8f){0,0,0,0,0,0,0,0}};
    #pragma unroll
    for (int kt = 0; kt < 2; ++kt) {
      #pragma unroll
      for (int c = 0; c < 2; ++c) {
        size_t rowK = (size_t)(b * Ss + kbase + kt * 16 + ql) * (3 * Ee)
                      + Ee + h * 64 + c * 32;
        Frag kf;
        kf.q[0] = *(const u32x4*)(qkv + rowK + khalf * 8);
        kf.q[1] = *(const u32x4*)(qkv + rowK + 16 + khalf * 8);
        st[kt] = wmma_bf16(kf, qf[c], st[kt]);
      }
    }

    // online softmax (per-lane = per-query; combine lane halves)
    float mx = -1e30f;
    #pragma unroll
    for (int r = 0; r < 8; ++r) {
      mx = fmaxf(mx, st[0][r]);
      mx = fmaxf(mx, st[1][r]);
    }
    mx *= scale;
    mx = fmaxf(mx, __shfl_xor(mx, 16));
    const float m_new = fmaxf(m_run, mx);
    const float alpha = __expf(m_run - m_new);
    float p0v[8], p1v[8], sum = 0.0f;
    #pragma unroll
    for (int r = 0; r < 8; ++r) {
      p0v[r] = __expf(st[0][r] * scale - m_new);
      p1v[r] = __expf(st[1][r] * scale - m_new);
      sum += p0v[r] + p1v[r];
    }
    sum += __shfl_xor(sum, 16);
    l_run = l_run * alpha + sum;
    m_run = m_new;
    #pragma unroll
    for (int d = 0; d < 4; ++d) oacc[d] = oacc[d] * alpha;

    u32x4 u0 = {pk2(p0v[0],p0v[1]), pk2(p0v[2],p0v[3]), pk2(p0v[4],p0v[5]), pk2(p0v[6],p0v[7])};
    u32x4 u1 = {pk2(p1v[0],p1v[1]), pk2(p1v[2],p1v[3]), pk2(p1v[4],p1v[5]), pk2(p1v[6],p1v[7])};
    *((u32x4*)&pMy[ql * 32 + 0 * 16 + khalf * 8]) = u0;
    *((u32x4*)&pMy[ql * 32 + 1 * 16 + khalf * 8]) = u1;
    wave_lds_sync();

    Frag pf;
    {
      const u32x4* pp = (const u32x4*)&pMy[ql * 32 + khalf * 8];
      pf.q[0] = pp[0];
      pf.q[1] = pp[2];
    }
    #pragma unroll
    for (int d = 0; d < 4; ++d) {
      Frag vf;
      vf.q[0] = *(const u32x4*)&vT[(d * 16 + ql) * 32 + khalf * 8];
      vf.q[1] = *(const u32x4*)&vT[(d * 16 + ql) * 32 + 16 + khalf * 8];
      oacc[d] = wmma_bf16(vf, pf, oacc[d]);
    }
  }

  const float inv = 1.0f / l_run;
  const size_t orow = (size_t)(b * Ss + qblk * 128 + wave * 16 + ql) * Ee + h * 64;
  #pragma unroll
  for (int d = 0; d < 4; ++d) {
    u32x4 ov = {pk2(oacc[d][0]*inv, oacc[d][1]*inv), pk2(oacc[d][2]*inv, oacc[d][3]*inv),
                pk2(oacc[d][4]*inv, oacc[d][5]*inv), pk2(oacc[d][6]*inv, oacc[d][7]*inv)};
    *(u32x4*)(o + orow + d * 16 + 8 * khalf) = ov;
  }
}

// ---------------------------------------------------------------------------
// out = LayerNorm(x + r) * g + be ; optional packed-bf16 second output.
// ---------------------------------------------------------------------------
__launch_bounds__(256)
__global__ void ln_fused(const float* __restrict__ x, const float* __restrict__ rs,
                         const float* __restrict__ g, const float* __restrict__ be,
                         float* __restrict__ out, u32* __restrict__ outb)
{
  __shared__ float red[8];
  const int row = blockIdx.x;
  const int tid = threadIdx.x;
  const size_t base = (size_t)row * Ee + tid * 4;

  f32x4 v = *(const f32x4*)(x + base);
  f32x4 r = *(const f32x4*)(rs + base);
  v.x += r.x; v.y += r.y; v.z += r.z; v.w += r.w;

  float s = v.x + v.y + v.z + v.w;
  #pragma unroll
  for (int off = 16; off; off >>= 1) s += __shfl_xor(s, off);
  if ((tid & 31) == 0) red[tid >> 5] = s;
  __syncthreads();
  float tot = 0.0f;
  #pragma unroll
  for (int i = 0; i < 8; ++i) tot += red[i];
  const float mu = tot * (1.0f / Ee);

  float dx = v.x - mu, dy = v.y - mu, dz = v.z - mu, dw = v.w - mu;
  float sq = dx*dx + dy*dy + dz*dz + dw*dw;
  __syncthreads();
  #pragma unroll
  for (int off = 16; off; off >>= 1) sq += __shfl_xor(sq, off);
  if ((tid & 31) == 0) red[tid >> 5] = sq;
  __syncthreads();
  float tv = 0.0f;
  #pragma unroll
  for (int i = 0; i < 8; ++i) tv += red[i];
  const float rstd = rsqrtf(tv * (1.0f / Ee) + 1e-5f);

  f32x4 gg = *(const f32x4*)(g + tid * 4);
  f32x4 bb = *(const f32x4*)(be + tid * 4);
  f32x4 ov = {dx * rstd * gg.x + bb.x, dy * rstd * gg.y + bb.y,
              dz * rstd * gg.z + bb.z, dw * rstd * gg.w + bb.w};
  *(f32x4*)(out + base) = ov;
  if (outb) {
    u32x2 pb = {pk2(ov.x, ov.y), pk2(ov.z, ov.w)};
    *(u32x2*)(outb + (size_t)row * (Ee / 2) + tid * 2) = pb;
  }
}

// ---------------------------------------------------------------------------
extern "C" void kernel_launch(void* const* d_in, const int* in_sizes, int n_in,
                              void* d_out, int out_size, void* d_ws, size_t ws_size,
                              hipStream_t stream)
{
  (void)in_sizes; (void)n_in; (void)out_size; (void)ws_size;
  const float* x    = (const float*)d_in[0];
  const float* wqkv = (const float*)d_in[1];
  const float* bqkv = (const float*)d_in[2];
  const float* wout = (const float*)d_in[3];
  const float* bout = (const float*)d_in[4];
  const float* w1   = (const float*)d_in[5];
  const float* b1   = (const float*)d_in[6];
  const float* w2   = (const float*)d_in[7];
  const float* b2   = (const float*)d_in[8];
  const float* g1   = (const float*)d_in[9];
  const float* be1  = (const float*)d_in[10];
  const float* g2   = (const float*)d_in[11];
  const float* be2  = (const float*)d_in[12];
  float* out = (float*)d_out;

  char* ws = (char*)d_ws;
  const size_t MB = 1ull << 20;
  // persistent bf16 weights
  u32* wqkv_bf = (u32*)(ws);             //  6 MB
  u32* wout_bf = (u32*)(ws + 6 * MB);    //  2 MB
  u32* w1_bf   = (u32*)(ws + 8 * MB);    //  8 MB
  u32* w2_bf   = (u32*)(ws + 16 * MB);   //  8 MB
  // activations
  u32* x_bf    = (u32*)(ws + 24 * MB);   // 16 MB
  u32* qkv_bf  = (u32*)(ws + 40 * MB);   // 48 MB (dead after attn)
  u32* o_bf    = (u32*)(ws + 88 * MB);   // 16 MB (dead after out-proj)
  float* tmpF  = (float*)(ws + 104 * MB);// 32 MB (attn_out, then ffn)
  float* ln1   = (float*)(ws + 136 * MB);// 32 MB
  u32* ln1_bf  = (u32*)(ws + 168 * MB);  // 16 MB
  u32* h_bf    = (u32*)(ws + 40 * MB);   // 64 MB (reuses qkv_bf + o_bf)

  const int M = Bb * Ss;
  dim3 blk(256);

  // 0) one-time f32 -> bf16 conversions
  f32_to_bf16<<<dim3(3 * Ee * Ee / 2048), blk, 0, stream>>>(wqkv, wqkv_bf);
  f32_to_bf16<<<dim3(Ee * Ee / 2048),     blk, 0, stream>>>(wout, wout_bf);
  f32_to_bf16<<<dim3(Ff * Ee / 2048),     blk, 0, stream>>>(w1, w1_bf);
  f32_to_bf16<<<dim3(Ee * Ff / 2048),     blk, 0, stream>>>(w2, w2_bf);
  f32_to_bf16<<<dim3(M * Ee / 2048),      blk, 0, stream>>>(x, x_bf);
  // 1) qkv = x @ w_qkv^T + b_qkv            (bf16 out)
  gemm_bf16<<<dim3(3 * Ee / 128, M / 128), blk, 0, stream>>>(
      x_bf, wqkv_bf, bqkv, nullptr, (u16*)qkv_bf, M, 3 * Ee, Ee, 0);
  // 2) o = attention(qkv)                   (bf16 out)
  attn_fwd<<<dim3(Bb * Hh, Ss / 128), blk, 0, stream>>>((const u16*)qkv_bf, (u16*)o_bf);
  // 3) attn_out = o @ w_out^T + b_out       (f32 out)
  gemm_bf16<<<dim3(Ee / 128, M / 128), blk, 0, stream>>>(
      o_bf, wout_bf, bout, tmpF, nullptr, M, Ee, Ee, 0);
  // 4) ln1 = LN(x + attn_out)               (f32 + bf16 out)
  ln_fused<<<dim3(M), blk, 0, stream>>>(x, tmpF, g1, be1, ln1, ln1_bf);
  // 5) h = relu(ln1 @ w1^T + b1)            (bf16 out)
  gemm_bf16<<<dim3(Ff / 128, M / 128), blk, 0, stream>>>(
      ln1_bf, w1_bf, b1, nullptr, (u16*)h_bf, M, Ff, Ee, 1);
  // 6) ffn = h @ w2^T + b2                  (f32 out)
  gemm_bf16<<<dim3(Ee / 128, M / 128), blk, 0, stream>>>(
      h_bf, w2_bf, b2, tmpF, nullptr, M, Ee, Ff, 0);
  // 7) out = LN(ln1 + ffn)
  ln_fused<<<dim3(M), blk, 0, stream>>>(ln1, tmpF, g2, be2, out, nullptr);
}
